// TorchDPOLoss_49246095016513
// MI455X (gfx1250) — matrix operand order repaired and checked by reading, not compiled, not run on hardware
//
#include <hip/hip_runtime.h>
#include <hip/hip_bf16.h>

// ---------------------------------------------------------------------------
// DPO loss, fused chunked linear-cross-entropy on gfx1250 (MI455X).
//   x:     (8,512,4096) fp32      y: (8,512) int
//   W:     (32000,4096) fp32      ref_W: (32000,4096) fp32
//   out:   scalar fp32 loss
// ---------------------------------------------------------------------------

#define HDIM  4096
#define VDIM  32000
#define NTOK  4096        // 8*512 tokens
#define SEQLEN 512
#define NSEQ  8
#define IGNORE_INDEX (-100)
#define BETA  0.1f

#define WAVES 8
#define TILES_PER_WAVE 5
#define BLOCK_COLS (WAVES * TILES_PER_WAVE * 16)   // 640 vocab cols / iteration
#define VCHUNKS 25
#define COLS_PER_CHUNK (VDIM / VCHUNKS)            // 1280
#define NITER (COLS_PER_CHUNK / BLOCK_COLS)        // 2
#define KCHUNKS (HDIM / 32)                        // 128

typedef __attribute__((ext_vector_type(16))) __bf16 v16bf;
typedef __attribute__((ext_vector_type(8)))  float  v8f;

// ---------------------------------------------------------------------------
// Kernel 1: logits = x @ Wm^T over one vocab chunk, online logsumexp partials.
// Block: 256 threads (8 waves), 16 tokens. The FULL 16x4096 x-panel is staged
// once in LDS as bf16 (128 KB of the WGP's 320 KB), so the GEMM inner loop has
// no barriers and no branches: ds_load_b128 A fragments + in-register
// fp32->bf16 W conversion + v_wmma_f32_16x16x32_bf16 (5 N-tiles per wave).
// ---------------------------------------------------------------------------
__global__ __launch_bounds__(256)
void dpo_lse_gemm_kernel(const float* __restrict__ x,
                         const float* __restrict__ Wmat,
                         const float* __restrict__ refW,
                         float* __restrict__ pmax,
                         float* __restrict__ psum)
{
    __shared__ __bf16 Atile[KCHUNKS][16][32];   // [k-chunk][token][k] : 128 KB
    __shared__ float  smax[WAVES][16];
    __shared__ float  ssum[WAVES][16];

    const int tid  = threadIdx.x;
    const int lane = tid & 31;
    const int wid  = tid >> 5;
    const int m    = lane & 15;             // row within half-wave
    const int hi   = lane >> 4;             // half-wave select

    const int m0    = blockIdx.y * 16;      // token tile base
    const int chunk = blockIdx.x;           // vocab chunk
    const int c0    = chunk * COLS_PER_CHUNK;
    const int model = blockIdx.z;
    const float* __restrict__ Wm = model ? refW : Wmat;

    // ---- one-time cooperative staging of the whole x panel (fp32 -> bf16) ----
    // 16*4096 elements, 2 per thread per sweep, coalesced float2 loads.
    for (int it = 0; it < (16 * HDIM) / (256 * 2); ++it) {
        const int e0   = (tid + it * 256) * 2;
        const int mrow = e0 >> 12;              // / HDIM
        const int k    = e0 & (HDIM - 1);
        const float2 v = *(const float2*)(x + (size_t)(m0 + mrow) * HDIM + k);
        Atile[k >> 5][mrow][k & 31]       = (__bf16)v.x;
        Atile[k >> 5][mrow][(k & 31) + 1] = (__bf16)v.y;
    }
    __syncthreads();

    // running logsumexp state: 8 tokens per half-wave (C-layout rows)
    float rmax[8], rsum[8];
#pragma unroll
    for (int r = 0; r < 8; ++r) { rmax[r] = -__builtin_inff(); rsum[r] = 0.f; }

    for (int it = 0; it < NITER; ++it) {
        const int nbase = c0 + it * BLOCK_COLS + wid * (TILES_PER_WAVE * 16);
        v8f acc[TILES_PER_WAVE];
#pragma unroll
        for (int t = 0; t < TILES_PER_WAVE; ++t) acc[t] = v8f{};

        // lane (hi,m): B column n = nbase + t*16 + m, K 8hi..8hi+7 / 16+8hi..23+8hi
        const float* __restrict__ wbase =
            Wm + (size_t)(nbase + m) * HDIM + hi * 8;

        for (int kc = 0; kc < KCHUNKS; ++kc) {
            // A fragment from LDS (ISA 16-bit A layout: lanes 0-15 K 0..7/16..23,
            // lanes 16-31 K 8..15/24..31)
            union { uint4 q[2]; v16bf v; } a;
            const uint4* ap = (const uint4*)&Atile[kc][m][hi * 8];
            a.q[0] = ap[0];     // 8 bf16
            a.q[1] = ap[2];     // +32 bytes -> next 8 bf16

#pragma unroll
            for (int t = 0; t < TILES_PER_WAVE; ++t) {
                union { v16bf v; __bf16 e[16]; } b;
                const float* wp = wbase + (size_t)t * 16 * HDIM + kc * 32;
#pragma unroll
                for (int i = 0; i < 16; ++i) b.e[i] = (__bf16)wp[i + (i & 8)];
                acc[t] = __builtin_amdgcn_wmma_f32_16x16x32_bf16(
                             false, a.v, false, b.v, (short)0, acc[t],
                             false, false);
            }
        }

        // online logsumexp update over the 80 columns this wave just produced
#pragma unroll
        for (int r = 0; r < 8; ++r) {
            float vm = acc[0][r];
#pragma unroll
            for (int t = 1; t < TILES_PER_WAVE; ++t) vm = fmaxf(vm, acc[t][r]);
            float mx = vm;
#pragma unroll
            for (int off = 8; off; off >>= 1)
                mx = fmaxf(mx, __shfl_xor(mx, off, 16));
            const float nm = fmaxf(rmax[r], mx);
            float s = 0.f;
#pragma unroll
            for (int t = 0; t < TILES_PER_WAVE; ++t)
                s += __expf(acc[t][r] - nm);
#pragma unroll
            for (int off = 8; off; off >>= 1)
                s += __shfl_xor(s, off, 16);
            rsum[r] = rsum[r] * __expf(rmax[r] - nm) + s;
            rmax[r] = nm;
        }
    }

    // combine the 8 waves' partials for the 16 tokens
    if ((lane & 15) == 0) {
#pragma unroll
        for (int r = 0; r < 8; ++r) {
            smax[wid][hi * 8 + r] = rmax[r];
            ssum[wid][hi * 8 + r] = rsum[r];
        }
    }
    __syncthreads();
    if (tid < 16) {
        float M = -__builtin_inff();
#pragma unroll
        for (int w = 0; w < WAVES; ++w) M = fmaxf(M, smax[w][tid]);
        float S = 0.f;
#pragma unroll
        for (int w = 0; w < WAVES; ++w) S += ssum[w][tid] * __expf(smax[w][tid] - M);
        const size_t idx = ((size_t)model * VCHUNKS + chunk) * NTOK + m0 + tid;
        pmax[idx] = M;
        psum[idx] = S;
    }
}

// ---------------------------------------------------------------------------
// Kernel 2: exact fp32 target logit  tok[model][t] = dot(x[t], Wm[y[t]]).
// One wave per token; coalesced float4 loads; wave shuffle reduction.
// ---------------------------------------------------------------------------
__global__ __launch_bounds__(256)
void dpo_target_logit_kernel(const float* __restrict__ x,
                             const float* __restrict__ Wmat,
                             const float* __restrict__ refW,
                             const int*   __restrict__ y,
                             float* __restrict__ tok)
{
    const int lane  = threadIdx.x & 31;
    const int wid   = threadIdx.x >> 5;
    const int token = blockIdx.x * 8 + wid;
    const int model = blockIdx.y;
    const float* __restrict__ Wm = model ? refW : Wmat;

    const int yv  = y[token];
    const int row = (yv == IGNORE_INDEX) ? 0 : yv;

    const float4* xp = (const float4*)(x  + (size_t)token * HDIM);
    const float4* wp = (const float4*)(Wm + (size_t)row   * HDIM);

    float acc = 0.f;
#pragma unroll 4
    for (int j = 0; j < HDIM / 128; ++j) {
        const float4 xv = xp[lane + 32 * j];
        const float4 wv = wp[lane + 32 * j];
        acc = fmaf(xv.x, wv.x, acc);
        acc = fmaf(xv.y, wv.y, acc);
        acc = fmaf(xv.z, wv.z, acc);
        acc = fmaf(xv.w, wv.w, acc);
    }
#pragma unroll
    for (int off = 16; off; off >>= 1) acc += __shfl_xor(acc, off, 32);
    if (lane == 0) tok[(size_t)model * NTOK + token] = acc;
}

// ---------------------------------------------------------------------------
// Kernel 3: combine partials -> logZ, per-token logp, per-sequence average,
// DPO sigmoid loss. Single block.
// ---------------------------------------------------------------------------
__global__ __launch_bounds__(256)
void dpo_finalize_kernel(const float* __restrict__ pmax,
                         const float* __restrict__ psum,
                         const float* __restrict__ tok,
                         const int*   __restrict__ y,
                         float* __restrict__ out)
{
    __shared__ float s_pol[NSEQ];
    __shared__ float s_ref[NSEQ];
    __shared__ int   s_cnt[NSEQ];

    const int tid = threadIdx.x;
    if (tid < NSEQ) { s_pol[tid] = 0.f; s_ref[tid] = 0.f; s_cnt[tid] = 0; }
    __syncthreads();

    for (int t = tid; t < NTOK; t += 256) {
        const int yv = y[t];
        if (yv == IGNORE_INDEX) continue;
        const int seq = t / SEQLEN;

        for (int model = 0; model < 2; ++model) {
            float pm[VCHUNKS], ps[VCHUNKS];
#pragma unroll
            for (int c = 0; c < VCHUNKS; ++c) {
                const size_t idx = ((size_t)model * VCHUNKS + c) * NTOK + t;
                pm[c] = pmax[idx];
                ps[c] = psum[idx];
            }
            float M = -__builtin_inff();
#pragma unroll
            for (int c = 0; c < VCHUNKS; ++c) M = fmaxf(M, pm[c]);
            float S = 0.f;
#pragma unroll
            for (int c = 0; c < VCHUNKS; ++c) S += ps[c] * __expf(pm[c] - M);
            const float logZ = M + __logf(S);
            const float lp   = tok[(size_t)model * NTOK + t] - logZ;
            atomicAdd(model ? &s_ref[seq] : &s_pol[seq], lp);
        }
        atomicAdd(&s_cnt[seq], 1);
    }
    __syncthreads();

    if (tid == 0) {
        float pol[NSEQ], ref[NSEQ];
#pragma unroll
        for (int s = 0; s < NSEQ; ++s) {
            const float inv = 1.f / (float)s_cnt[s];
            pol[s] = s_pol[s] * inv;
            ref[s] = s_ref[s] * inv;
        }
        float acc = 0.f;
#pragma unroll
        for (int i = 0; i < NSEQ / 2; ++i) {
            const float d = BETA * ((pol[i] - ref[i]) - (pol[i + 4] - ref[i + 4]));
            const float z = -d;  // loss = softplus(-d), numerically stable
            acc += fmaxf(z, 0.f) + log1pf(__expf(-fabsf(z)));
        }
        out[0] = acc * (2.0f / (float)NSEQ);
    }
}

// ---------------------------------------------------------------------------
extern "C" void kernel_launch(void* const* d_in, const int* in_sizes, int n_in,
                              void* d_out, int out_size, void* d_ws, size_t ws_size,
                              hipStream_t stream)
{
    const float* x    = (const float*)d_in[0];
    const int*   y    = (const int*)  d_in[1];
    const float* W    = (const float*)d_in[2];
    const float* refW = (const float*)d_in[3];
    float* out = (float*)d_out;

    float* ws   = (float*)d_ws;
    float* pmax = ws;                                   // 2*25*4096 floats
    float* psum = pmax + (size_t)2 * VCHUNKS * NTOK;    // 2*25*4096 floats
    float* tok  = psum + (size_t)2 * VCHUNKS * NTOK;    // 2*4096 floats
    (void)in_sizes; (void)n_in; (void)out_size; (void)ws_size;

    dpo_lse_gemm_kernel<<<dim3(VCHUNKS, NTOK / 16, 2), 256, 0, stream>>>(
        x, W, refW, pmax, psum);
    dpo_target_logit_kernel<<<dim3(NTOK / 8, 2), 256, 0, stream>>>(
        x, W, refW, y, tok);
    dpo_finalize_kernel<<<1, 256, 0, stream>>>(pmax, psum, tok, y, out);
}
